// RNN_15092515078459
// MI455X (gfx1250) — matrix operand order repaired and checked by reading, not compiled, not run on hardware
//
#include <hip/hip_runtime.h>

// ---------------------------------------------------------------------------
// GRU language model for MI455X (gfx1250, wave32, WMMA 16x16x32 bf16).
// (1) transpose/convert weights to bf16 [N,K] layout,
// (2) parallel x-part gate preactivations for all B*T tokens (WMMA),
// (3) single-workgroup sequential GRU: h state in registers, recurrent
//     weights preloaded into per-wave registers, only A-staging in LDS,
// (4) store-bound LM head with nontemporal fp32 stores (WMMA).
// ---------------------------------------------------------------------------

typedef __attribute__((ext_vector_type(16))) __bf16         v16bf;
typedef __attribute__((ext_vector_type(8)))  float          v8f;
typedef __attribute__((ext_vector_type(8)))  unsigned short v8us;
typedef __attribute__((ext_vector_type(16))) unsigned short v16us;

#define BB 16
#define TT 256
#define VV 32000
#define EE 64
#define HH 128
#define BT 4096           // BB*TT
#define LDSTRIDE 136      // padded bf16 row stride (272B) to spread LDS banks

// ---- ws layout (bytes) ----
#define WLMT_OFF   0u          // [VV][HH] bf16 : 8,192,000
#define WXT_OFF    8192000u    // [3][HH][EE] bf16 : 49,152
#define WHT_OFF    8241152u    // [3][HH][HH] bf16 : 98,304
#define XPRE_OFF   8339456u    // [3][BT][HH] f32 : 6,291,456
#define HID_OFF    14630912u   // [BT][HH] bf16 : 1,048,576

// Native f32->bf16 (RNE) via hardware cvt; storage stays u16.
static __device__ __forceinline__ unsigned short f2bf(float f) {
  __bf16 b = (__bf16)f;
  return __builtin_bit_cast(unsigned short, b);
}

static __device__ __forceinline__ v8f zero8() {
  v8f z;
#pragma unroll
  for (int i = 0; i < 8; ++i) z[i] = 0.0f;
  return z;
}

static __device__ __forceinline__ v8f wmma_bf16(v16us a_us, v16us b_us, v8f c) {
  v16bf a = __builtin_bit_cast(v16bf, a_us);
  v16bf b = __builtin_bit_cast(v16bf, b_us);
  return __builtin_amdgcn_wmma_f32_16x16x32_bf16(false, a, false, b, (short)0, c,
                                                 false, false);
}

// A-operand (16x32 bf16): lane holds row m; per K32 step the lane needs
// K = {k0+8*half .. +7} in elems 0..7 and K = {k0+16+8*half .. +7} in 8..15.
static __device__ __forceinline__ v16us load_a_chunks(const unsigned short* row,
                                                      int k0, int half) {
  v8us lo = *(const v8us*)(row + k0 + 8 * half);
  v8us hi = *(const v8us*)(row + k0 + 16 + 8 * half);
  v16us r;
#pragma unroll
  for (int j = 0; j < 8; ++j) { r[j] = lo[j]; r[8 + j] = hi[j]; }
  return r;
}

// B-operand (32x16 bf16): lane holds column n; elems 0..7 = K {k0+8*half..+7},
// elems 8..15 = K {k0+16+8*half..+7} (B stored transposed [N][K]).
static __device__ __forceinline__ v16us load_b_chunks(const unsigned short* colrow,
                                                      int k0, int half) {
  v8us lo = *(const v8us*)(colrow + k0 + 8 * half);
  v8us hi = *(const v8us*)(colrow + k0 + 16 + 8 * half);
  v16us r;
#pragma unroll
  for (int j = 0; j < 8; ++j) { r[j] = lo[j]; r[8 + j] = hi[j]; }
  return r;
}

// ---------------------------------------------------------------------------
// Kernel 1a: tiled transpose+convert Wlm [HH,VV] f32 -> WlmT [VV,HH] bf16.
__global__ __launch_bounds__(256) void k_transpose_wlm(
    const float* __restrict__ Wlm, unsigned short* __restrict__ WlmT) {
  __shared__ float tile[32][33];
  int tx = threadIdx.x & 31, ty = threadIdx.x >> 5;  // 32x8
  int v0 = blockIdx.x * 32, h0 = blockIdx.y * 32;
#pragma unroll
  for (int j = 0; j < 32; j += 8)
    tile[ty + j][tx] = Wlm[(h0 + ty + j) * VV + (v0 + tx)];
  __syncthreads();
#pragma unroll
  for (int j = 0; j < 32; j += 8)
    WlmT[(v0 + ty + j) * HH + (h0 + tx)] = f2bf(tile[tx][ty + j]);
}

// Kernel 1b: gate weights Wg [(EE+HH),HH] f32 -> WxT [HH][EE], WhT [HH][HH] bf16.
__global__ __launch_bounds__(256) void k_convert_gates(
    const float* __restrict__ Wz, const float* __restrict__ Wr,
    const float* __restrict__ Wh, unsigned short* __restrict__ WxT,
    unsigned short* __restrict__ WhT) {
  int i = blockIdx.x * 256 + threadIdx.x;
  const int per_gate = HH * EE + HH * HH;
  if (i >= 3 * per_gate) return;
  int g = i / per_gate, rle = i % per_gate;
  const float* W = (g == 0) ? Wz : ((g == 1) ? Wr : Wh);
  if (rle < HH * EE) {
    int n = rle / EE, k = rle % EE;
    WxT[g * HH * EE + n * EE + k] = f2bf(W[k * HH + n]);
  } else {
    int r2 = rle - HH * EE, n = r2 / HH, k = r2 % HH;
    WhT[g * HH * HH + n * HH + k] = f2bf(W[(EE + k) * HH + n]);
  }
}

// ---------------------------------------------------------------------------
// Kernel 2: x-part gate preactivations for all flattened rows (b*T+t):
// xpre[g][row][n] = bias_g[n] + emb[idx[row]] @ Wgx.  WMMA, K=64 (2 steps).
__global__ __launch_bounds__(256) void k_xpre(
    const int* __restrict__ idx, const float* __restrict__ emb,
    const unsigned short* __restrict__ WxT, const float* __restrict__ bz,
    const float* __restrict__ br, const float* __restrict__ bh,
    float* __restrict__ xpre) {
  int lane = threadIdx.x & 31, wave = threadIdx.x >> 5;
  int m = lane & 15, half = lane >> 4;
  int mtile = blockIdx.x;
  int row = mtile * 16 + m;
  const float* e = emb + (long)idx[row] * EE;

  v16us a[2];
#pragma unroll
  for (int ks = 0; ks < 2; ++ks) {
    int k0 = 32 * ks;
#pragma unroll
    for (int j = 0; j < 8; ++j) {
      a[ks][j]     = f2bf(e[k0 + 8 * half + j]);
      a[ks][8 + j] = f2bf(e[k0 + 16 + 8 * half + j]);
    }
  }

  int col = wave * 16 + (lane & 15);
  const float* biases[3] = {bz, br, bh};
#pragma unroll
  for (int g = 0; g < 3; ++g) {
    const unsigned short* wt = WxT + g * HH * EE + col * EE;
    v8f c = zero8();
    c = wmma_bf16(a[0], load_b_chunks(wt, 0, half), c);
    c = wmma_bf16(a[1], load_b_chunks(wt, 32, half), c);
    float bias = biases[g][col];
    float* outp = xpre + (long)g * BT * HH;
#pragma unroll
    for (int i = 0; i < 8; ++i)
      outp[(long)(mtile * 16 + i + 8 * half) * HH + col] = c[i] + bias;
  }
}

// ---------------------------------------------------------------------------
// Kernel 3: sequential GRU. One workgroup, 8 waves; wave w owns gate columns
// [16w,16w+16). The fp32 state lives in registers (each lane always owns the
// same (row,col) cells); recurrent weights are preloaded into per-wave
// registers (3 gates x 4 k-steps x 8 VGPRs = 96 VGPRs); LDS holds only the
// bf16 A-staging tiles (h and r*h), so each step is LDS->WMMA with 2 barriers.
__global__ __launch_bounds__(256) void k_gru(
    const float* __restrict__ start, const unsigned short* __restrict__ WhT,
    const float* __restrict__ xpre, unsigned short* __restrict__ hidden) {
  __shared__ unsigned short hA[BB * LDSTRIDE];     // bf16 A staging (h)
  __shared__ unsigned short rhA[BB * LDSTRIDE];    // bf16 A staging (r*h)

  int tid = threadIdx.x;
  int lane = tid & 31;
  int m = lane & 15, half = lane >> 4;
  int col = (tid >> 5) * 16 + (lane & 15);

  // Preload recurrent weights (B-operands) for this wave's columns.
  v16us wz[4], wr[4], wh[4];
#pragma unroll
  for (int ks = 0; ks < 4; ++ks) {
    wz[ks] = load_b_chunks(WhT + 0 * HH * HH + col * HH, 32 * ks, half);
    wr[ks] = load_b_chunks(WhT + 1 * HH * HH + col * HH, 32 * ks, half);
    wh[ks] = load_b_chunks(WhT + 2 * HH * HH + col * HH, 32 * ks, half);
  }

  // Init state: h0 = start (broadcast over batch rows). Lane owns rows
  // rb = i + 8*half (i = 0..7) at column `col`.
  float hold[8];
  float h0 = start[col];
  unsigned short h0b = f2bf(h0);
#pragma unroll
  for (int i = 0; i < 8; ++i) {
    hold[i] = h0;
    hA[(i + 8 * half) * LDSTRIDE + col] = h0b;
  }
  __syncthreads();

  const float* xz = xpre;
  const float* xr = xpre + (long)BT * HH;
  const float* xh = xpre + 2l * BT * HH;

  for (int t = 0; t < TT; ++t) {
    // --- phase 1: r and z gates ---
    v16us a0 = load_a_chunks(hA + m * LDSTRIDE, 0, half);
    v16us a1 = load_a_chunks(hA + m * LDSTRIDE, 32, half);
    v16us a2 = load_a_chunks(hA + m * LDSTRIDE, 64, half);
    v16us a3 = load_a_chunks(hA + m * LDSTRIDE, 96, half);

    v8f cz, cr;
#pragma unroll
    for (int i = 0; i < 8; ++i) {
      int rb = i + 8 * half;
      cz[i] = xz[(long)(rb * TT + t) * HH + col];
      cr[i] = xr[(long)(rb * TT + t) * HH + col];
    }
    cz = wmma_bf16(a0, wz[0], cz);
    cz = wmma_bf16(a1, wz[1], cz);
    cz = wmma_bf16(a2, wz[2], cz);
    cz = wmma_bf16(a3, wz[3], cz);
    cr = wmma_bf16(a0, wr[0], cr);
    cr = wmma_bf16(a1, wr[1], cr);
    cr = wmma_bf16(a2, wr[2], cr);
    cr = wmma_bf16(a3, wr[3], cr);

    float zf[8];
#pragma unroll
    for (int i = 0; i < 8; ++i) {
      int rb = i + 8 * half;
      zf[i] = 1.0f / (1.0f + __expf(-cz[i]));
      float rg = 1.0f / (1.0f + __expf(-cr[i]));
      rhA[rb * LDSTRIDE + col] = f2bf(rg * hold[i]);
    }
    __syncthreads();

    // --- phase 2: candidate hbar and state update ---
    v16us g0 = load_a_chunks(rhA + m * LDSTRIDE, 0, half);
    v16us g1 = load_a_chunks(rhA + m * LDSTRIDE, 32, half);
    v16us g2 = load_a_chunks(rhA + m * LDSTRIDE, 64, half);
    v16us g3 = load_a_chunks(rhA + m * LDSTRIDE, 96, half);
    v8f ch;
#pragma unroll
    for (int i = 0; i < 8; ++i)
      ch[i] = xh[(long)((i + 8 * half) * TT + t) * HH + col];
    ch = wmma_bf16(g0, wh[0], ch);
    ch = wmma_bf16(g1, wh[1], ch);
    ch = wmma_bf16(g2, wh[2], ch);
    ch = wmma_bf16(g3, wh[3], ch);

#pragma unroll
    for (int i = 0; i < 8; ++i) {
      int rb = i + 8 * half;
      float hb = tanhf(ch[i]);
      float hn = hold[i] + zf[i] * (hb - hold[i]);  // (1-z)h + z*hbar
      hold[i] = hn;
      unsigned short hnb = f2bf(hn);
      hA[rb * LDSTRIDE + col] = hnb;                 // A for next step
      hidden[(long)(rb * TT + t) * HH + col] = hnb;  // LM-head input
    }
    __syncthreads();
  }
}

// ---------------------------------------------------------------------------
// Kernel 4: LM head. Each wave: 4 M-tiles x 1 N-tile (reuses each B-chunk for
// 4 WMMAs to cut L2 traffic on WlmT). Nontemporal fp32 stores (524 MB, HBM-
// store bound at ~22.5us @ 23.3 TB/s).
__global__ __launch_bounds__(256) void k_lmhead(
    const unsigned short* __restrict__ hidden,
    const unsigned short* __restrict__ WlmT, const float* __restrict__ blm,
    float* __restrict__ out) {
  int lane = threadIdx.x & 31, wave = threadIdx.x >> 5;
  int m = lane & 15, half = lane >> 4;
  int ntile = blockIdx.x * 8 + wave;            // 0..1999
  int mgrp = blockIdx.y;                        // 0..63 (4 M-tiles each)
  int col = ntile * 16 + (lane & 15);
  const unsigned short* bcol = WlmT + (long)col * HH;

  v8f acc[4];
#pragma unroll
  for (int q = 0; q < 4; ++q) acc[q] = zero8();

#pragma unroll
  for (int ks = 0; ks < 4; ++ks) {
    int k0 = 32 * ks;
    v16us b = load_b_chunks(bcol, k0, half);
#pragma unroll
    for (int q = 0; q < 4; ++q) {
      const unsigned short* arow =
          hidden + (long)(mgrp * 64 + q * 16 + m) * HH;
      v16us a = load_a_chunks(arow, k0, half);
      acc[q] = wmma_bf16(a, b, acc[q]);
    }
  }

  float bias = blm[col];
#pragma unroll
  for (int q = 0; q < 4; ++q) {
#pragma unroll
    for (int i = 0; i < 8; ++i) {
      long row = mgrp * 64 + q * 16 + i + 8 * half;
      __builtin_nontemporal_store(acc[q][i] + bias, out + row * VV + col);
    }
  }
}

// ---------------------------------------------------------------------------
extern "C" void kernel_launch(void* const* d_in, const int* in_sizes, int n_in,
                              void* d_out, int out_size, void* d_ws,
                              size_t ws_size, hipStream_t stream) {
  const int*   idx   = (const int*)d_in[0];
  const float* start = (const float*)d_in[1];
  const float* emb   = (const float*)d_in[2];
  const float* Wz    = (const float*)d_in[3];
  const float* bz    = (const float*)d_in[4];
  const float* Wr    = (const float*)d_in[5];
  const float* br    = (const float*)d_in[6];
  const float* Wh    = (const float*)d_in[7];
  const float* bh    = (const float*)d_in[8];
  const float* Wlm   = (const float*)d_in[9];
  const float* blm   = (const float*)d_in[10];
  float* out = (float*)d_out;

  char* ws = (char*)d_ws;
  unsigned short* WlmT = (unsigned short*)(ws + WLMT_OFF);
  unsigned short* WxT  = (unsigned short*)(ws + WXT_OFF);
  unsigned short* WhT  = (unsigned short*)(ws + WHT_OFF);
  float*          xpre = (float*)(ws + XPRE_OFF);
  unsigned short* hid  = (unsigned short*)(ws + HID_OFF);

  k_transpose_wlm<<<dim3(VV / 32, HH / 32), 256, 0, stream>>>(Wlm, WlmT);
  int gate_elems = 3 * (HH * EE + HH * HH);
  k_convert_gates<<<(gate_elems + 255) / 256, 256, 0, stream>>>(Wz, Wr, Wh,
                                                                WxT, WhT);
  k_xpre<<<BT / 16, 256, 0, stream>>>(idx, emb, WxT, bz, br, bh, xpre);
  k_gru<<<1, 256, 0, stream>>>(start, WhT, xpre, hid);
  k_lmhead<<<dim3((VV / 16) / 8, (BT / 16) / 4), 256, 0, stream>>>(hid, WlmT,
                                                                   blm, out);
}
